// Net_18949395710668
// MI455X (gfx1250) — compile-verified, matrix operand order checked by reading, not run on hardware
//
#include <hip/hip_runtime.h>
#include <hip/hip_bf16.h>
#include <math.h>

#define N_NODES 16000
#define NGRAPH  16
#define NPER    1000
#define HDIM    64
#define KNN     20
#define DIN     8
#define LAYERS  4
#define BN_EPS  1e-5f
#define NTILES_COL 63   // ceil(1000/16)
#define LPAD    68      // LDS row pad: 272B rows -> 16B aligned, stride%64banks==4

typedef float  v2f  __attribute__((ext_vector_type(2)));
typedef float  v8f  __attribute__((ext_vector_type(8)));
typedef __bf16 v16bf __attribute__((ext_vector_type(16)));

union BF16x16 { unsigned u[8]; v16bf v; };

__device__ __forceinline__ unsigned short f2bf(float f) {
  unsigned u = __float_as_uint(f);
  unsigned r = ((u >> 16) & 1u) + 0x7FFFu;   // round-to-nearest-even
  return (unsigned short)((u + r) >> 16);
}
__device__ __forceinline__ unsigned packbf(float a, float b) {
  return (unsigned)f2bf(a) | ((unsigned)f2bf(b) << 16);
}
__device__ __forceinline__ float elu1(float x) { return x > 0.f ? x : (__expf(x) - 1.f); }

// CDNA5 async global->LDS copy (ASYNCcnt path), ISA §15.18 opcode 98.
// vdst = LDS byte address (low 32 bits of flat shared address), vaddr = 64b global.
__device__ __forceinline__ void async_ld_b128(const void* gptr, void* lptr) {
  asm volatile("global_load_async_to_lds_b128 %0, %1, off"
               :: "v"((unsigned)(size_t)lptr), "v"(gptr) : "memory");
}
__device__ __forceinline__ void wait_async0() {
  asm volatile("s_wait_asynccnt 0" ::: "memory");
}

// ---------------------------------------------------------------- lc_encode
__global__ void lc_encode_kernel(const float* __restrict__ x,
    const float* __restrict__ W1, const float* __restrict__ b1,
    const float* __restrict__ W2, const float* __restrict__ b2,
    float* __restrict__ h) {
  int i = blockIdx.x * blockDim.x + threadIdx.x;
  if (i >= N_NODES) return;
  float xi[DIN];
#pragma unroll
  for (int d = 0; d < DIN; ++d) xi[d] = x[i * DIN + d];
  float hid[32];
#pragma unroll
  for (int j = 0; j < 32; ++j) {
    float s = b1[j];
#pragma unroll
    for (int d = 0; d < DIN; ++d) s += xi[d] * W1[d * 32 + j];
    hid[j] = elu1(s);
  }
  for (int o = 0; o < HDIM; ++o) {
    float s = b2[o];
#pragma unroll
    for (int j = 0; j < 32; ++j) s += hid[j] * W2[j * HDIM + o];
    h[i * HDIM + o] = elu1(s);
  }
}

// ------------------------------------------- pack edge-MLP weights to bf16 B-layout
// bpack[((ntile*4+kc)*32 + lane)*8 + v] : V_WMMA_F32_16X16X32_BF16 B-operand layout
__global__ void pack_w_kernel(const float* __restrict__ W /*128x64*/,
                              unsigned* __restrict__ bp) {
  int nt = blockIdx.x >> 2, kc = blockIdx.x & 3;
  int lane = threadIdx.x;
  int n = nt * 16 + (lane & 15);
  int khalf = (lane >> 4) << 3;
#pragma unroll
  for (int v = 0; v < 8; ++v) {
    int kl = (v < 4) ? (khalf + 2 * v) : (16 + khalf + 2 * (v - 4));
    int k = kc * 32 + kl;
    bp[((nt * 4 + kc) * 32 + lane) * 8 + v] =
        packbf(W[k * 64 + n], W[(k + 1) * 64 + n]);
  }
}

// ---------------------------------------------------------------- kNN (per layer)
// One wave per (graph, 16-row block). Gram tile via fp32 WMMA; top-20 in VGPRs.
// Tiles staged with async global->LDS b128 copies.
__global__ void __launch_bounds__(32) knn_kernel(const float* __restrict__ h,
                                                 int* __restrict__ nbr) {
  __shared__ float hq[16][LPAD];
  __shared__ float hc[16][LPAD];
  __shared__ float dt[16][17];
  __shared__ float csq[16];
  int g = blockIdx.x / NTILES_COL;
  int rt = blockIdx.x % NTILES_COL;
  int row0 = rt * 16;
  int base = g * NPER;
  int lane = threadIdx.x;

  // stage query tile: 16 rows x 64 floats = 256 float4 chunks
#pragma unroll
  for (int t0 = 0; t0 < 256; t0 += 32) {
    int t = t0 + lane;
    int r = t >> 4, c4 = (t & 15) << 2;
    int row = row0 + r; if (row >= NPER) row = NPER - 1;
    async_ld_b128(&h[(base + row) * HDIM + c4], &hq[r][c4]);
  }
  wait_async0();
  __syncthreads();

  float bd[KNN]; int bi[KNN];
#pragma unroll
  for (int s = 0; s < KNN; ++s) { bd[s] = 3.4e38f; bi[s] = base; }
  float curmax = 3.4e38f; int maxslot = 0;

  int mr = lane & 15;
  int koff = (lane >> 4) << 1;   // K sub-offset 0 or 2 (fp32 16x16x4 A/B layout)

  for (int ct = 0; ct < NTILES_COL; ++ct) {
    int col0 = ct * 16;
    // prefetch rows of the next two tiles into the hierarchy
    { int nc = col0 + 16 + lane; if (nc >= NPER) nc = NPER - 1;
      __builtin_prefetch(&h[(base + nc) * HDIM], 0, 0); }
    // stage column tile via async copies
#pragma unroll
    for (int t0 = 0; t0 < 256; t0 += 32) {
      int t = t0 + lane;
      int r = t >> 4, c4 = (t & 15) << 2;
      int col = col0 + r; if (col >= NPER) col = NPER - 1;
      async_ld_b128(&h[(base + col) * HDIM + c4], &hc[r][c4]);
    }
    wait_async0();
    __syncthreads();

    if (lane < 16) {
      float s = 0.f;
      for (int d = 0; d < HDIM; ++d) { float vv = hc[lane][d]; s += vv * vv; }
      csq[lane] = s;
    }
    __syncthreads();

    v8f acc = {};
#pragma unroll
    for (int kk = 0; kk < 16; ++kk) {          // K=64 in chunks of 4
      int kb = kk * 4 + koff;
      v2f a, b;
      a.x = hq[mr][kb]; a.y = hq[mr][kb + 1];
      b.x = hc[mr][kb]; b.y = hc[mr][kb + 1];
      acc = __builtin_amdgcn_wmma_f32_16x16x4_f32(
          false, a, false, b, (short)0, acc, false, false);
    }
    // d2 (up to a per-row constant) = |h_j|^2 - 2*G  -> LDS tile
    {
      int col = lane & 15;
      int rbase = (lane >> 4) << 3;
      float cs = csq[col];
#pragma unroll
      for (int r = 0; r < 8; ++r)
        dt[rbase + r][col] = cs - 2.0f * acc[r];
    }
    __syncthreads();

    if (lane < 16 && (row0 + lane) < NPER) {
#pragma unroll
      for (int c = 0; c < 16; ++c) {
        int j = col0 + c;
        float d = dt[lane][c];
        if (j < NPER && d < curmax) {
#pragma unroll
          for (int s = 0; s < KNN; ++s)
            if (s == maxslot) { bd[s] = d; bi[s] = base + j; }
          curmax = -3.4e38f;
#pragma unroll
          for (int s = 0; s < KNN; ++s)
            if (bd[s] > curmax) { curmax = bd[s]; maxslot = s; }
        }
      }
    }
    __syncthreads();
  }

  if (lane < 16 && (row0 + lane) < NPER) {
    int node = base + row0 + lane;
#pragma unroll
    for (int s = 0; s < KNN; ++s) nbr[node * KNN + s] = bi[s];
  }
}

// ----------------------------------------- edge MLP + max-aggr + residual
// One wave per node; 2 M-tiles (20 edges padded with duplicates of edge 0;
// duplicates are exact under max-aggregation). bf16 WMMA with fp32 accum.
__global__ void __launch_bounds__(32) edge_kernel(const float* __restrict__ h,
    const int* __restrict__ nbr, const unsigned* __restrict__ bp,
    const float* __restrict__ bias,
    const float* __restrict__ bng, const float* __restrict__ bnb,
    const float* __restrict__ bnm, const float* __restrict__ bnv,
    float* __restrict__ hout) {
  __shared__ float hi[64];
  __shared__ float hj[16][LPAD];
  __shared__ int   nb[KNN];
  int i = blockIdx.x;
  int lane = threadIdx.x;
  hi[lane]      = h[i * HDIM + lane];
  hi[lane + 32] = h[i * HDIM + lane + 32];
  if (lane < KNN) nb[lane] = nbr[i * KNN + lane];
  __syncthreads();

  int m = lane & 15;
  int khalf = (lane >> 4) << 3;
  float rm[4] = {-3.4e38f, -3.4e38f, -3.4e38f, -3.4e38f};

  for (int mt = 0; mt < 2; ++mt) {
    // stage 16 neighbor rows via async global->LDS copies
#pragma unroll
    for (int t0 = 0; t0 < 256; t0 += 32) {
      int t = t0 + lane;
      int r = t >> 4, c4 = (t & 15) << 2;
      int s = mt * 16 + r;
      int j = (s < KNN) ? nb[s] : nb[0];
      async_ld_b128(&h[j * HDIM + c4], &hj[r][c4]);
    }
    wait_async0();
    __syncthreads();

    // A operands (bf16 16x16x32 layout): e = [h_i , h_j - h_i], K=128 in 4 chunks
    BF16x16 A[4];
#pragma unroll
    for (int kc = 0; kc < 4; ++kc) {
#pragma unroll
      for (int v = 0; v < 8; ++v) {
        int kl = (v < 4) ? (khalf + 2 * v) : (16 + khalf + 2 * (v - 4));
        int k = kc * 32 + kl;
        float f0, f1;
        if (k < 64) { f0 = hi[k]; f1 = hi[k + 1]; }
        else        { f0 = hj[m][k - 64] - hi[k - 64];
                      f1 = hj[m][k - 63] - hi[k - 63]; }
        A[kc].u[v] = packbf(f0, f1);
      }
    }

#pragma unroll
    for (int nt = 0; nt < 4; ++nt) {
      v8f acc = {};
#pragma unroll
      for (int kc = 0; kc < 4; ++kc) {
        BF16x16 B;
        const uint4* p = (const uint4*)&bp[((nt * 4 + kc) * 32 + lane) * 8];
        uint4 q0 = p[0], q1 = p[1];
        B.u[0] = q0.x; B.u[1] = q0.y; B.u[2] = q0.z; B.u[3] = q0.w;
        B.u[4] = q1.x; B.u[5] = q1.y; B.u[6] = q1.z; B.u[7] = q1.w;
        acc = __builtin_amdgcn_wmma_f32_16x16x32_bf16(
            false, A[kc].v, false, B.v, (short)0, acc, false, false);
      }
      int col = nt * 16 + m;
      float bb = bias[col], gg = bng[col], be = bnb[col], mm = bnm[col];
      float inv = rsqrtf(bnv[col] + BN_EPS);
#pragma unroll
      for (int r = 0; r < 8; ++r) {
        float msg = elu1(acc[r] + bb);
        msg = gg * (msg - mm) * inv + be;
        rm[nt] = fmaxf(rm[nt], msg);
      }
    }
    __syncthreads();
  }

#pragma unroll
  for (int nt = 0; nt < 4; ++nt) {
    float vv = rm[nt];
    vv = fmaxf(vv, __shfl_xor(vv, 16, 32));   // combine M halves (wave32)
    if (lane < 16)
      hout[i * HDIM + nt * 16 + lane] = vv + hi[nt * 16 + lane];
  }
}

// ---------------------------------------------------------------- heads
__global__ void heads_kernel(const float* __restrict__ h,
    const float* oW1, const float* ob1, const float* oW2, const float* ob2,
    const float* oW3, const float* ob3,
    const float* sW1, const float* sb1, const float* sW2, const float* sb2,
    const float* sW3, const float* sb3,
    float* __restrict__ out) {
  __shared__ float hs[64];
  __shared__ float s1[64];
  __shared__ float s2[32];
  int i = blockIdx.x;
  int t = threadIdx.x;
  hs[t] = h[i * HDIM + t];
  __syncthreads();
  // contrastive head 64->64->32->8
  {
    float a = ob1[t];
    for (int k = 0; k < 64; ++k) a += hs[k] * oW1[k * 64 + t];
    s1[t] = elu1(a);
    __syncthreads();
    if (t < 32) {
      float a2 = ob2[t];
      for (int k = 0; k < 64; ++k) a2 += s1[k] * oW2[k * 32 + t];
      s2[t] = elu1(a2);
    }
    __syncthreads();
    if (t < 8) {
      float a3 = ob3[t];
      for (int k = 0; k < 32; ++k) a3 += s2[k] * oW3[k * 8 + t];
      out[i * 8 + t] = a3;
    }
    __syncthreads();
  }
  // split head 64->64->32->1
  {
    float a = sb1[t];
    for (int k = 0; k < 64; ++k) a += hs[k] * sW1[k * 64 + t];
    s1[t] = elu1(a);
    __syncthreads();
    if (t < 32) {
      float a2 = sb2[t];
      for (int k = 0; k < 64; ++k) a2 += s1[k] * sW2[k * 32 + t];
      s2[t] = elu1(a2);
    }
    __syncthreads();
    if (t == 0) {
      float a3 = sb3[0];
      for (int k = 0; k < 32; ++k) a3 += s2[k] * sW3[k];
      out[N_NODES * 8 + i] = a3;
    }
  }
}

__global__ void batch_kernel(float* __restrict__ out) {
  int i = blockIdx.x * blockDim.x + threadIdx.x;
  if (i < N_NODES) out[N_NODES * 8 + N_NODES + i] = (float)(i / NPER);
}

// ---------------------------------------------------------------- launch
extern "C" void kernel_launch(void* const* d_in, const int* in_sizes, int n_in,
                              void* d_out, int out_size, void* d_ws, size_t ws_size,
                              hipStream_t stream) {
  (void)in_sizes; (void)n_in; (void)out_size; (void)ws_size;
  const float* x     = (const float*)d_in[0];
  const float* lcW1  = (const float*)d_in[2];
  const float* lcb1  = (const float*)d_in[3];
  const float* lcW2  = (const float*)d_in[4];
  const float* lcb2  = (const float*)d_in[5];
  const float* convW = (const float*)d_in[6];
  const float* convb = (const float*)d_in[7];
  const float* bng   = (const float*)d_in[8];
  const float* bnb   = (const float*)d_in[9];
  const float* bnm   = (const float*)d_in[10];
  const float* bnv   = (const float*)d_in[11];
  const float* oW1   = (const float*)d_in[12];
  const float* ob1   = (const float*)d_in[13];
  const float* oW2   = (const float*)d_in[14];
  const float* ob2   = (const float*)d_in[15];
  const float* oW3   = (const float*)d_in[16];
  const float* ob3   = (const float*)d_in[17];
  const float* sW1   = (const float*)d_in[18];
  const float* sb1   = (const float*)d_in[19];
  const float* sW2   = (const float*)d_in[20];
  const float* sb2   = (const float*)d_in[21];
  const float* sW3   = (const float*)d_in[22];
  const float* sb3   = (const float*)d_in[23];
  float* out = (float*)d_out;

  char* ws = (char*)d_ws;
  float*    hA  = (float*)(ws);
  float*    hB  = (float*)(ws + 4096000);
  int*      nbr = (int*)  (ws + 8192000);
  unsigned* bp  = (unsigned*)(ws + 9472000);

  lc_encode_kernel<<<(N_NODES + 127) / 128, 128, 0, stream>>>(
      x, lcW1, lcb1, lcW2, lcb2, hA);

  float* hcur = hA; float* hnext = hB;
  for (int l = 0; l < LAYERS; ++l) {
    pack_w_kernel<<<16, 32, 0, stream>>>(convW + l * 128 * 64, bp);
    knn_kernel<<<NGRAPH * NTILES_COL, 32, 0, stream>>>(hcur, nbr);
    edge_kernel<<<N_NODES, 32, 0, stream>>>(hcur, nbr, bp, convb + l * 64,
        bng + l * 64, bnb + l * 64, bnm + l * 64, bnv + l * 64, hnext);
    float* tmp = hcur; hcur = hnext; hnext = tmp;
  }

  heads_kernel<<<N_NODES, 64, 0, stream>>>(hcur,
      oW1, ob1, oW2, ob2, oW3, ob3, sW1, sb1, sW2, sb2, sW3, sb3, out);
  batch_kernel<<<(N_NODES + 255) / 256, 256, 0, stream>>>(out);
}
